// GNet_16252156248485
// MI455X (gfx1250) — compile-verified, matrix operand order checked by reading, not run on hardware
//
#include <hip/hip_runtime.h>

#define N_NODES 16384
#define E_EDGES 131072
#define D 768
#define NT 4
#define NB 3
#define K2 (2*D)

typedef __attribute__((ext_vector_type(16))) __bf16 v16bf;
typedef __attribute__((ext_vector_type(8)))  float  v8f;
typedef __attribute__((ext_vector_type(4)))  int    v4i;

#if defined(__has_builtin)
#if __has_builtin(__builtin_amdgcn_global_load_async_to_lds_b128)
#define HAVE_ASYNC_LDS 1
#endif
#if __has_builtin(__builtin_amdgcn_s_wait_asynccnt)
#define HAVE_WAIT_ASYNC 1
#endif
#endif

#define AS1 __attribute__((address_space(1)))
#define AS3 __attribute__((address_space(3)))

__device__ __forceinline__ void wait_async0() {
#if defined(HAVE_WAIT_ASYNC)
    __builtin_amdgcn_s_wait_asynccnt(0);
#else
    asm volatile("s_wait_asynccnt 0x0" ::: "memory");
#endif
}

__device__ __forceinline__ unsigned short f2bf(float f) {
    unsigned int u = __builtin_bit_cast(unsigned int, f);
    unsigned int r = u + 0x7FFFu + ((u >> 16) & 1u);
    return (unsigned short)(r >> 16);
}
__device__ __forceinline__ float bf2f(unsigned short h) {
    unsigned int u = ((unsigned int)h) << 16;
    return __builtin_bit_cast(float, u);
}

__device__ __forceinline__ v16bf ld_frag(const unsigned short* p, int off0, int off1) {
    union { uint4 q[2]; v16bf v; } u;
    u.q[0] = *reinterpret_cast<const uint4*>(p + off0);
    u.q[1] = *reinterpret_cast<const uint4*>(p + off1);
    return u.v;
}

// ---------------- LayerNorm: h [N,768] f32 -> x bf16 ----------------
__global__ __launch_bounds__(256) void k_layernorm(const float* __restrict__ h,
                                                   const float* __restrict__ gamma,
                                                   const float* __restrict__ beta,
                                                   unsigned short* __restrict__ xout) {
    const int row = blockIdx.x, tid = threadIdx.x;
    const float* hr = h + (size_t)row * D;
    float v0 = hr[tid], v1 = hr[tid + 256], v2 = hr[tid + 512];
    __shared__ float red[256];
    red[tid] = v0 + v1 + v2;
    __syncthreads();
    for (int s = 128; s > 0; s >>= 1) { if (tid < s) red[tid] += red[tid + s]; __syncthreads(); }
    float mu = red[0] * (1.0f / D);
    __syncthreads();
    float d0 = v0 - mu, d1 = v1 - mu, d2 = v2 - mu;
    red[tid] = d0 * d0 + d1 * d1 + d2 * d2;
    __syncthreads();
    for (int s = 128; s > 0; s >>= 1) { if (tid < s) red[tid] += red[tid + s]; __syncthreads(); }
    float rs = rsqrtf(red[0] * (1.0f / D) + 1e-5f);
    unsigned short* xo = xout + (size_t)row * D;
    xo[tid]       = f2bf(d0 * rs * gamma[tid]       + beta[tid]);
    xo[tid + 256] = f2bf(d1 * rs * gamma[tid + 256] + beta[tid + 256]);
    xo[tid + 512] = f2bf(d2 * rs * gamma[tid + 512] + beta[tid + 512]);
}

// ---------------- Fold basis weights: W_f[t] = sum_b coeff[t,b]*W[b], -> bf16 ----------------
__global__ __launch_bounds__(256) void k_fold(const float* __restrict__ W0, const float* __restrict__ c0,
                                              const float* __restrict__ W1, const float* __restrict__ c1,
                                              const float* __restrict__ W2, const float* __restrict__ c2,
                                              unsigned short* __restrict__ wfold) {
    const size_t idx = (size_t)blockIdx.x * 256 + threadIdx.x;   // 12*D*D total
    const size_t per = (size_t)NT * D * D;
    int f = (int)(idx / per);
    size_t r = idx % per;
    int t = (int)(r / ((size_t)D * D));
    size_t kc = r % ((size_t)D * D);
    const float* W = (f == 0) ? W0 : (f == 1) ? W1 : W2;
    const float* c = (f == 0) ? c0 : (f == 1) ? c1 : c2;
    float acc = 0.f;
    #pragma unroll
    for (int b = 0; b < NB; ++b) acc += c[t * NB + b] * W[(size_t)b * D * D + kc];
    wfold[idx] = f2bf(acc);
}

__global__ __launch_bounds__(256) void k_cvt_wlin(const float* __restrict__ w, unsigned short* __restrict__ o) {
    size_t i = (size_t)blockIdx.x * 256 + threadIdx.x;   // 1536*768
    o[i] = f2bf(w[i]);
}

__global__ __launch_bounds__(256) void k_zero_i32(int* __restrict__ p) {
    p[(size_t)blockIdx.x * 256 + threadIdx.x] = 0;
}

__global__ __launch_bounds__(256) void k_count(const int* __restrict__ dst, const int* __restrict__ etype,
                                               int* __restrict__ cnt) {
    int e = blockIdx.x * 256 + threadIdx.x;
    atomicAdd(&cnt[dst[e] * NT + etype[e]], 1);
}

// nB[v,d] = sum_t cnt[v,t]*relu(Hd[t,v,d]); also zero the feat half of nodeAcc
__global__ __launch_bounds__(256) void k_dstsum(const int* __restrict__ cnt,
                                                const unsigned short* __restrict__ tabHd,
                                                float* __restrict__ nacc) {
    size_t idx = (size_t)blockIdx.x * 256 + threadIdx.x;  // N*D
    int v = (int)(idx / D), d = (int)(idx % D);
    float a = 0.f;
    #pragma unroll
    for (int t = 0; t < NT; ++t)
        a += (float)cnt[v * NT + t] * bf2f(tabHd[((size_t)t * N_NODES + v) * D + d]);
    nacc[(size_t)v * K2 + d] = 0.f;
    nacc[(size_t)v * K2 + D + d] = a;
}

// ---------------- WMMA GEMM: table = relu(x @ Wfold[z]) stored bf16 ----------------
// block: 256 thr (8 waves), block tile 128(M) x 128(N), wave tile 32x64, K-step 32
__global__ __launch_bounds__(256) void k_gemm_tab(const unsigned short* __restrict__ X,
                                                  const unsigned short* __restrict__ Bfam,
                                                  unsigned short* __restrict__ Ctab) {
    __shared__ alignas(16) unsigned short sA[128 * 40];
    __shared__ alignas(16) unsigned short sB[128 * 40];   // transposed: [col][k]
    const int tid = threadIdx.x;
    const int lane = tid & 31, wave = tid >> 5;
    const int wm = wave & 3, wn = wave >> 2;
    const int half = lane >> 4, lrow = lane & 15;
    const unsigned short* Bmat = Bfam + (size_t)blockIdx.z * D * D;
    unsigned short* Cmat = Ctab + (size_t)blockIdx.z * N_NODES * D;
    const int m0 = blockIdx.x * 128, n0 = blockIdx.y * 128;
    const int ar = tid >> 2, ac = (tid & 3) * 8;

    v8f acc[2][4] = {};
    for (int kt = 0; kt < D; kt += 32) {
        // ---- stage A (128x32 bf16): async global->LDS if available ----
#if defined(HAVE_ASYNC_LDS)
        __builtin_amdgcn_global_load_async_to_lds_b128(
            (AS1 v4i*)(X + (size_t)(m0 + ar) * D + kt + ac),
            (AS3 v4i*)(sA + ar * 40 + ac), 0, 0);
        __builtin_amdgcn_global_load_async_to_lds_b128(
            (AS1 v4i*)(X + (size_t)(m0 + ar + 64) * D + kt + ac),
            (AS3 v4i*)(sA + (ar + 64) * 40 + ac), 0, 0);
#else
        *(uint4*)(sA + ar * 40 + ac)        = *(const uint4*)(X + (size_t)(m0 + ar) * D + kt + ac);
        *(uint4*)(sA + (ar + 64) * 40 + ac) = *(const uint4*)(X + (size_t)(m0 + ar + 64) * D + kt + ac);
#endif
        // ---- stage B (32x128 bf16) transposed into LDS ----
        #pragma unroll
        for (int i = 0; i < 2; ++i) {
            int li = tid * 2 + i;
            int k = li >> 4, c = (li & 15) * 8;
            uint4 q = *(const uint4*)(Bmat + (size_t)(kt + k) * D + n0 + c);
            const unsigned short* e = (const unsigned short*)&q;
            #pragma unroll
            for (int j = 0; j < 8; ++j) sB[(c + j) * 40 + k] = e[j];
        }
#if defined(HAVE_ASYNC_LDS)
        wait_async0();
#endif
        __syncthreads();
        v16bf afr[2], bfr[4];
        #pragma unroll
        for (int mi = 0; mi < 2; ++mi)
            afr[mi] = ld_frag(sA + (wm * 32 + mi * 16 + lrow) * 40, half * 8, 16 + half * 8);
        #pragma unroll
        for (int ni = 0; ni < 4; ++ni)
            bfr[ni] = ld_frag(sB + (wn * 64 + ni * 16 + lrow) * 40, half * 8, 16 + half * 8);
        #pragma unroll
        for (int mi = 0; mi < 2; ++mi)
            #pragma unroll
            for (int ni = 0; ni < 4; ++ni)
                acc[mi][ni] = __builtin_amdgcn_wmma_f32_16x16x32_bf16(
                    false, afr[mi], false, bfr[ni], (short)0, acc[mi][ni], false, false);
        __syncthreads();
    }
    #pragma unroll
    for (int mi = 0; mi < 2; ++mi) {
        #pragma unroll
        for (int ni = 0; ni < 4; ++ni) {
            int col = n0 + wn * 64 + ni * 16 + lrow;
            int mb = m0 + wm * 32 + mi * 16 + half * 8;
            #pragma unroll
            for (int r = 0; r < 8; ++r) {
                float v = acc[mi][ni][r];
                v = v > 0.f ? v : 0.f;
                Cmat[(size_t)(mb + r) * D + col] = f2bf(v);
            }
        }
    }
}

// ---------------- perm: stable hete-first argsort ----------------
__global__ __launch_bounds__(256) void k_cnthete(const int* __restrict__ hete, int* __restrict__ bcnt) {
    __shared__ int s;
    int tid = threadIdx.x;
    if (tid == 0) s = 0;
    __syncthreads();
    int f = hete[blockIdx.x * 256 + tid] > 0 ? 1 : 0;
    atomicAdd(&s, f);
    __syncthreads();
    if (tid == 0) bcnt[blockIdx.x] = s;
}

__global__ __launch_bounds__(512) void k_scan(int* __restrict__ bcnt) {
    __shared__ int a[512], b[512];
    int tid = threadIdx.x;
    int v = bcnt[tid];
    a[tid] = v;
    __syncthreads();
    int* cur = a; int* nxt = b;
    for (int off = 1; off < 512; off <<= 1) {
        int x = cur[tid] + (tid >= off ? cur[tid - off] : 0);
        nxt[tid] = x;
        __syncthreads();
        int* t = cur; cur = nxt; nxt = t;
    }
    bcnt[512 + tid] = cur[tid] - v;             // exclusive prefix
    if (tid == 511) bcnt[1024] = cur[511];      // total hete
}

__global__ __launch_bounds__(256) void k_perm(const int* __restrict__ hete, const int* __restrict__ bcnt,
                                              int* __restrict__ perm) {
    __shared__ int a[256], b[256];
    int tid = threadIdx.x, blk = blockIdx.x;
    int e = blk * 256 + tid;
    int f = hete[e] > 0 ? 1 : 0;
    a[tid] = f;
    __syncthreads();
    int* cur = a; int* nxt = b;
    for (int off = 1; off < 256; off <<= 1) {
        int x = cur[tid] + (tid >= off ? cur[tid - off] : 0);
        nxt[tid] = x;
        __syncthreads();
        int* t = cur; cur = nxt; nxt = t;
    }
    int hb = cur[tid] - f;
    int ex = bcnt[512 + blk], nH = bcnt[1024];
    int pos = f ? (ex + hb) : (nH + (blk * 256 - ex) + (tid - hb));
    perm[pos] = e;
}

// ---------------- edge scatter: one wave per edge position ----------------
__global__ __launch_bounds__(256) void k_scatter(const int* __restrict__ perm, const int* __restrict__ src,
                                                 const int* __restrict__ dst, const int* __restrict__ etype,
                                                 const int* __restrict__ hete,
                                                 const unsigned short* __restrict__ tabT,
                                                 const unsigned short* __restrict__ tabU,
                                                 float* __restrict__ nacc) {
    const int tid = threadIdx.x, lane = tid & 31, wave = tid >> 5;
    const int ep = blockIdx.x * 8 + wave;
    const int f = perm[ep];
    const int t = etype[f], s = src[f];
    const unsigned short* row = (hete[f] > 0 ? tabT : tabU) + ((size_t)t * N_NODES + s) * D;
    float* o = nacc + (size_t)dst[ep] * K2;
    #pragma unroll
    for (int j = 0; j < 24; ++j) {
        int c = lane + j * 32;
        unsafeAtomicAdd(o + c, bf2f(row[c]));
    }
}

// ---------------- final GEMM: z = relu(nacc @ Wlin + b), f32 out ----------------
// block tile 128x128, wave tile 32x64, K-step 32, K=1536
__global__ __launch_bounds__(256) void k_gemm_out(const float* __restrict__ A,
                                                  const unsigned short* __restrict__ Bm,
                                                  const float* __restrict__ bias,
                                                  float* __restrict__ out) {
    __shared__ alignas(16) unsigned short sA[128 * 40];
    __shared__ alignas(16) unsigned short sB[128 * 40];
    const int tid = threadIdx.x;
    const int lane = tid & 31, wave = tid >> 5;
    const int wm = wave & 3, wn = wave >> 2;
    const int half = lane >> 4, lrow = lane & 15;
    const int m0 = blockIdx.x * 128, n0 = blockIdx.y * 128;

    v8f acc[2][4] = {};
    for (int kt = 0; kt < K2; kt += 32) {
        #pragma unroll
        for (int i = 0; i < 4; ++i) {
            int li = tid + i * 256;
            int row = li >> 3, c4 = (li & 7) * 4;
            float4 fv = *(const float4*)(A + (size_t)(m0 + row) * K2 + kt + c4);
            ushort4 u;
            u.x = f2bf(fv.x); u.y = f2bf(fv.y); u.z = f2bf(fv.z); u.w = f2bf(fv.w);
            *(ushort4*)(sA + row * 40 + c4) = u;
        }
        #pragma unroll
        for (int i = 0; i < 2; ++i) {
            int li = tid * 2 + i;
            int k = li >> 4, c = (li & 15) * 8;
            uint4 q = *(const uint4*)(Bm + (size_t)(kt + k) * D + n0 + c);
            const unsigned short* e = (const unsigned short*)&q;
            #pragma unroll
            for (int j = 0; j < 8; ++j) sB[(c + j) * 40 + k] = e[j];
        }
        __syncthreads();
        v16bf afr[2], bfr[4];
        #pragma unroll
        for (int mi = 0; mi < 2; ++mi)
            afr[mi] = ld_frag(sA + (wm * 32 + mi * 16 + lrow) * 40, half * 8, 16 + half * 8);
        #pragma unroll
        for (int ni = 0; ni < 4; ++ni)
            bfr[ni] = ld_frag(sB + (wn * 64 + ni * 16 + lrow) * 40, half * 8, 16 + half * 8);
        #pragma unroll
        for (int mi = 0; mi < 2; ++mi)
            #pragma unroll
            for (int ni = 0; ni < 4; ++ni)
                acc[mi][ni] = __builtin_amdgcn_wmma_f32_16x16x32_bf16(
                    false, afr[mi], false, bfr[ni], (short)0, acc[mi][ni], false, false);
        __syncthreads();
    }
    #pragma unroll
    for (int mi = 0; mi < 2; ++mi) {
        #pragma unroll
        for (int ni = 0; ni < 4; ++ni) {
            int col = n0 + wn * 64 + ni * 16 + lrow;
            int mb = m0 + wm * 32 + mi * 16 + half * 8;
            float bv = bias[col];
            #pragma unroll
            for (int r = 0; r < 8; ++r) {
                float v = acc[mi][ni][r] + bv;
                out[(size_t)(mb + r) * D + col] = v > 0.f ? v : 0.f;
            }
        }
    }
}

// ---------------- row L2 normalize in place ----------------
__global__ __launch_bounds__(256) void k_norm(float* __restrict__ z) {
    const int row = blockIdx.x, tid = threadIdx.x;
    float* zr = z + (size_t)row * D;
    float v0 = zr[tid], v1 = zr[tid + 256], v2 = zr[tid + 512];
    __shared__ float red[256];
    red[tid] = v0 * v0 + v1 * v1 + v2 * v2;
    __syncthreads();
    for (int s = 128; s > 0; s >>= 1) { if (tid < s) red[tid] += red[tid + s]; __syncthreads(); }
    float n = sqrtf(red[0]);
    if (n == 0.f) n = 1.f;
    float inv = 1.0f / n;
    zr[tid] = v0 * inv; zr[tid + 256] = v1 * inv; zr[tid + 512] = v2 * inv;
}

extern "C" void kernel_launch(void* const* d_in, const int* in_sizes, int n_in,
                              void* d_out, int out_size, void* d_ws, size_t ws_size,
                              hipStream_t stream) {
    const float* h        = (const float*)d_in[0];
    const float* W_text   = (const float*)d_in[1];
    const float* c_text   = (const float*)d_in[2];
    const float* W_user   = (const float*)d_in[3];
    const float* c_user   = (const float*)d_in[4];
    const float* W_dst    = (const float*)d_in[5];
    const float* c_dst    = (const float*)d_in[6];
    const float* ln_gamma = (const float*)d_in[7];
    const float* ln_beta  = (const float*)d_in[8];
    const float* W_lin    = (const float*)d_in[9];
    const float* b_lin    = (const float*)d_in[10];
    const int*   src      = (const int*)d_in[11];
    const int*   dst      = (const int*)d_in[12];
    const int*   etype    = (const int*)d_in[13];
    const int*   hete     = (const int*)d_in[14];
    float* out = (float*)d_out;

    char* ws = (char*)d_ws;
    size_t off = 0;
    unsigned short* x_bf   = (unsigned short*)(ws + off); off += (size_t)N_NODES * D * 2;          // 25.2 MB
    unsigned short* wfold  = (unsigned short*)(ws + off); off += (size_t)3 * NT * D * D * 2;       // 14.2 MB
    unsigned short* wlin_b = (unsigned short*)(ws + off); off += (size_t)K2 * D * 2;               // 2.4 MB
    unsigned short* tab0   = (unsigned short*)(ws + off); off += (size_t)NT * N_NODES * D * 2;     // 100.7 MB
    unsigned short* tab1   = (unsigned short*)(ws + off); off += (size_t)NT * N_NODES * D * 2;     // 100.7 MB
    float*          nacc   = (float*)(ws + off);          off += (size_t)N_NODES * K2 * 4;         // 100.7 MB
    int*            cnt    = (int*)(ws + off);            off += (size_t)N_NODES * NT * 4;
    int*            perm   = (int*)(ws + off);            off += (size_t)E_EDGES * 4;
    int*            bcnt   = (int*)(ws + off);            off += 4352;

    // 1. LayerNorm -> x bf16
    k_layernorm<<<N_NODES, 256, 0, stream>>>(h, ln_gamma, ln_beta, x_bf);
    // 2. fold weights (12 D x D mats) + convert W_lin
    k_fold<<<(3 * NT * D * D) / 256, 256, 0, stream>>>(W_text, c_text, W_user, c_user, W_dst, c_dst, wfold);
    k_cvt_wlin<<<(K2 * D) / 256, 256, 0, stream>>>(W_lin, wlin_b);
    // 3. dst type counts
    k_zero_i32<<<(N_NODES * NT) / 256, 256, 0, stream>>>(cnt);
    k_count<<<E_EDGES / 256, 256, 0, stream>>>(dst, etype, cnt);
    // 4. Hd table (slot 0), then fold into nacc second half (and zero first half)
    k_gemm_tab<<<dim3(128, 6, 4), 256, 0, stream>>>(x_bf, wfold + (size_t)2 * NT * D * D, tab0);
    k_dstsum<<<((size_t)N_NODES * D) / 256, 256, 0, stream>>>(cnt, tab0, nacc);
    // 5. Ht (slot 0, Hd consumed) and Hu (slot 1)
    k_gemm_tab<<<dim3(128, 6, 4), 256, 0, stream>>>(x_bf, wfold, tab0);
    k_gemm_tab<<<dim3(128, 6, 4), 256, 0, stream>>>(x_bf, wfold + (size_t)1 * NT * D * D, tab1);
    // 6. stable hete-first permutation
    k_cnthete<<<E_EDGES / 256, 256, 0, stream>>>(hete, bcnt);
    k_scan<<<1, 512, 0, stream>>>(bcnt);
    k_perm<<<E_EDGES / 256, 256, 0, stream>>>(hete, bcnt, perm);
    // 7. edge gather + atomic scatter into nacc first half
    k_scatter<<<E_EDGES / 8, 256, 0, stream>>>(perm, src, dst, etype, hete, tab0, tab1, nacc);
    // 8. final linear + relu
    k_gemm_out<<<dim3(128, 6), 256, 0, stream>>>(nacc, wlin_b, b_lin, out);
    // 9. row L2 normalize
    k_norm<<<N_NODES, 256, 0, stream>>>(out);
}
